// PairedKidneyBackbone_6055903888058
// MI455X (gfx1250) — matrix operand order repaired and checked by reading, not compiled, not run on hardware
//
#include <hip/hip_runtime.h>
#include <hip/hip_bf16.h>
#include <math.h>

typedef __attribute__((ext_vector_type(16))) _Float16 v16h;
typedef __attribute__((ext_vector_type(8)))  _Float16 v8h;
typedef __attribute__((ext_vector_type(8)))  float    v8f;

__device__ __forceinline__ v16h cat16(v8h lo, v8h hi) {
  return __builtin_shufflevector(lo, hi, 0,1,2,3,4,5,6,7,8,9,10,11,12,13,14,15);
}

// ---------------------------------------------------------------------------
// Generic WMMA GEMM:  C[M,F] = act(A16[M,K] @ Wt16[F,K]^T + bias) (+res) (*rowscale)
// A row-major [M,K] f16, Wt row-major [F,K] f16 (i.e. B pre-transposed).
// 256 threads = 8 waves; block tile 128x128; wave tile 64(M)x32(F); K-step 32.
// ---------------------------------------------------------------------------
__global__ __launch_bounds__(256) void gemm_wmma_f16(
    const _Float16* __restrict__ A, const _Float16* __restrict__ Wt,
    const float* __restrict__ bias, const float* __restrict__ res,
    const float* __restrict__ rowscale,
    float* __restrict__ out32, _Float16* __restrict__ out16,
    int M, int F, int K,
    long long sA, long long sW, long long sC, int relu)
{
  __shared__ __align__(16) _Float16 As[2][128][40];
  __shared__ __align__(16) _Float16 Bs[2][128][40];

  const int tid  = threadIdx.x;
  const int wid  = tid >> 5;
  const int lane = tid & 31;
  const int l16  = lane & 15;
  const int half = lane >> 4;          // lane half selects K sub-range
  const int wm   = wid >> 2;           // 0..1  (wave M index)
  const int wf   = wid & 3;            // 0..3  (wave F index)
  const int bm   = blockIdx.x * 128;
  const int bf   = blockIdx.y * 128;
  const int bz   = blockIdx.z;

  A  += (long long)bz * sA;
  Wt += (long long)bz * sW;
  const long long cbase = (long long)bz * sC;

  // each thread stages two 8-half (16B) chunks per tile
  const int row0 = tid >> 2;           // 0..63
  const int row1 = row0 + 64;          // 64..127
  const int col0 = (tid & 3) * 8;      // 0,8,16,24

  v8f acc[4][2];
  #pragma unroll
  for (int mi = 0; mi < 4; ++mi)
    #pragma unroll
    for (int fi = 0; fi < 2; ++fi) { v8f z = {}; acc[mi][fi] = z; }

  const int KT = K >> 5;

  // prologue: load + stage tile 0
  v8h ra0 = *(const v8h*)(A  + (long long)(bm + row0) * K + col0);
  v8h ra1 = *(const v8h*)(A  + (long long)(bm + row1) * K + col0);
  v8h rb0 = *(const v8h*)(Wt + (long long)(bf + row0) * K + col0);
  v8h rb1 = *(const v8h*)(Wt + (long long)(bf + row1) * K + col0);
  *(v8h*)&As[0][row0][col0] = ra0;
  *(v8h*)&As[0][row1][col0] = ra1;
  *(v8h*)&Bs[0][row0][col0] = rb0;
  *(v8h*)&Bs[0][row1][col0] = rb1;

  for (int kt = 0; kt < KT; ++kt) {
    const int cur = kt & 1;
    if (kt + 1 < KT) {
      const int k0 = (kt + 1) << 5;
      ra0 = *(const v8h*)(A  + (long long)(bm + row0) * K + k0 + col0);
      ra1 = *(const v8h*)(A  + (long long)(bm + row1) * K + k0 + col0);
      rb0 = *(const v8h*)(Wt + (long long)(bf + row0) * K + k0 + col0);
      rb1 = *(const v8h*)(Wt + (long long)(bf + row1) * K + k0 + col0);
      if (kt + 2 < KT) {  // gfx1250 global_prefetch_b8 on the tile after next
        const int k2 = (kt + 2) << 5;
        __builtin_prefetch(A  + (long long)(bm + row0) * K + k2 + col0, 0, 1);
        __builtin_prefetch(Wt + (long long)(bf + row0) * K + k2 + col0, 0, 1);
      }
    }
    __syncthreads();

    // fragment loads per ISA 16-bit layouts (lanes 0-15 / 16-31 K-split)
    v16h afr[4], bfr[2];
    #pragma unroll
    for (int mi = 0; mi < 4; ++mi) {
      const _Float16* p = &As[cur][wm * 64 + mi * 16 + l16][0];
      afr[mi] = cat16(*(const v8h*)(p + half * 8),
                      *(const v8h*)(p + 16 + half * 8));
    }
    #pragma unroll
    for (int fi = 0; fi < 2; ++fi) {
      const _Float16* p = &Bs[cur][wf * 32 + fi * 16 + l16][0];
      bfr[fi] = cat16(*(const v8h*)(p + half * 8),
                      *(const v8h*)(p + 16 + half * 8));
    }
    #pragma unroll
    for (int mi = 0; mi < 4; ++mi)
      #pragma unroll
      for (int fi = 0; fi < 2; ++fi)
        acc[mi][fi] = __builtin_amdgcn_wmma_f32_16x16x32_f16(
            false, afr[mi], false, bfr[fi], (short)0, acc[mi][fi], false, false);

    if (kt + 1 < KT) {
      const int nb = cur ^ 1;
      *(v8h*)&As[nb][row0][col0] = ra0;
      *(v8h*)&As[nb][row1][col0] = ra1;
      *(v8h*)&Bs[nb][row0][col0] = rb0;
      *(v8h*)&Bs[nb][row1][col0] = rb1;
    }
  }

  // epilogue: C layout — lane<16: n=lane, m=r; lane>=16: n=lane-16, m=8+r
  #pragma unroll
  for (int mi = 0; mi < 4; ++mi) {
    #pragma unroll
    for (int fi = 0; fi < 2; ++fi) {
      const int n = bf + wf * 32 + fi * 16 + l16;
      const float bv = bias ? bias[n] : 0.f;
      #pragma unroll
      for (int r = 0; r < 8; ++r) {
        const int m = bm + wm * 64 + mi * 16 + half * 8 + r;
        float v = acc[mi][fi][r] + bv;
        if (relu) v = fmaxf(v, 0.f);
        const long long idx = cbase + (long long)m * F + n;
        if (res)      v += res[idx];
        if (rowscale) v *= rowscale[(long long)bz * M + m];
        if (out32) out32[idx] = v;
        if (out16) out16[idx] = (_Float16)v;
      }
    }
  }
}

// ---------------------------------------------------------------------------
// Tiled transpose f32 [R,C] -> f16 [C,R]  (also used to pre-transpose weights)
// ---------------------------------------------------------------------------
__global__ __launch_bounds__(256) void transpose_f32_to_f16(
    const float* __restrict__ in, _Float16* __restrict__ out,
    int R, int C, long long inStride, long long outStride)
{
  const int b = blockIdx.z;
  in  += (long long)b * inStride;
  out += (long long)b * outStride;
  __shared__ float tile[32][33];
  const int c0 = blockIdx.x * 32, r0 = blockIdx.y * 32;
  const int tx = threadIdx.x, ty = threadIdx.y;   // (32,8)
  #pragma unroll
  for (int i = 0; i < 4; ++i)
    tile[ty + i * 8][tx] = in[(long long)(r0 + ty + i * 8) * C + c0 + tx];
  __syncthreads();
  #pragma unroll
  for (int i = 0; i < 4; ++i)
    out[(long long)(c0 + ty + i * 8) * R + r0 + tx] = (_Float16)tile[tx][ty + i * 8];
}

// adjacency [b,s,t] int32 -> adjT [b,t,s] u8 (mask bit), tiled transpose
__global__ __launch_bounds__(256) void adjT_kernel(
    const int* __restrict__ adj, unsigned char* __restrict__ adjT, int N)
{
  const int b = blockIdx.z;
  const int* in = adj + (long long)b * N * N;
  unsigned char* out = adjT + (long long)b * N * N;
  __shared__ unsigned char tile[32][33];
  const int t0 = blockIdx.x * 32, s0 = blockIdx.y * 32;
  const int tx = threadIdx.x, ty = threadIdx.y;
  #pragma unroll
  for (int i = 0; i < 4; ++i)
    tile[ty + i * 8][tx] =
        (unsigned char)(in[(long long)(s0 + ty + i * 8) * N + t0 + tx] != 0);
  __syncthreads();
  #pragma unroll
  for (int i = 0; i < 4; ++i)
    out[(long long)(t0 + ty + i * 8) * N + s0 + tx] = tile[tx][ty + i * 8];
}

// ---------------------------------------------------------------------------
// features + first embedding linear (K=6): writes x1 f16 [M,512]
// ---------------------------------------------------------------------------
__global__ __launch_bounds__(256) void feat_emb1_kernel(
    const float* __restrict__ arrivals, const float* __restrict__ departures,
    const float* __restrict__ ihm, const float* __restrict__ ts,
    const float* __restrict__ tt, const float* __restrict__ W1,
    const float* __restrict__ b1, _Float16* __restrict__ x1, int N, int H)
{
  const long long m = blockIdx.x;        // b*N + n
  const int b = (int)(m / N);
  const float a = arrivals[m], d = departures[m];
  const float tsb = ts[b];
  const float f0 = (tsb - a) / (d - a);
  const float f1 = tsb / tt[b];
  const float f2 = fmodf(f1, 2.f), f3 = fmodf(f1, 4.f), f4 = fmodf(f1, 7.f);
  const float f5 = ihm[m];
  for (int h = threadIdx.x; h < H; h += blockDim.x) {
    float v = b1[h] + f0 * W1[h] + f1 * W1[H + h] + f2 * W1[2 * H + h]
            + f3 * W1[3 * H + h] + f4 * W1[4 * H + h] + f5 * W1[5 * H + h];
    x1[m * H + h] = (_Float16)v;
  }
}

// e_src/e_dst per row: one wave per row, shuffle reduce
__global__ __launch_bounds__(256) void edot_kernel(
    const float* __restrict__ hw, const float* __restrict__ asrc,
    const float* __restrict__ adst, float* __restrict__ esrc,
    float* __restrict__ edst, int F)
{
  const int w = threadIdx.x >> 5, lane = threadIdx.x & 31;
  const long long row = (long long)blockIdx.x * 8 + w;
  const float* r = hw + row * F;
  float s1 = 0.f, s2 = 0.f;
  for (int f = lane; f < F; f += 32) {
    const float v = r[f];
    s1 += v * asrc[f];
    s2 += v * adst[f];
  }
  #pragma unroll
  for (int o = 16; o; o >>= 1) {
    s1 += __shfl_xor(s1, o, 32);
    s2 += __shfl_xor(s2, o, 32);
  }
  if (!lane) { esrc[row] = s1; edst[row] = s2; }
}

// masked leaky-relu softmax over sources per target row; writes alpha f16
__global__ __launch_bounds__(256) void alpha_kernel(
    const unsigned char* __restrict__ adjT, const float* __restrict__ esrc,
    const float* __restrict__ edst, _Float16* __restrict__ alpha, int N)
{
  const long long row = blockIdx.x;             // b*N + t
  const int b = (int)(row / N);
  const int t = (int)(row - (long long)b * N);
  const unsigned char* arow = adjT + row * N;
  const float* es = esrc + (long long)b * N;
  const float ed = edst[row];
  const int tid = threadIdx.x, w = tid >> 5, lane = tid & 31;
  __shared__ float red[8];

  float lg[8];
  float mx = -3.4e38f;
  #pragma unroll
  for (int j = 0; j < 8; ++j) {
    const int s = j * 256 + tid;
    float l = ed + es[s];
    l = l > 0.f ? l : 0.2f * l;
    const bool on = (arow[s] != 0) || (s == t);
    lg[j] = on ? l : -INFINITY;
    mx = fmaxf(mx, lg[j]);
  }
  #pragma unroll
  for (int o = 16; o; o >>= 1) mx = fmaxf(mx, __shfl_xor(mx, o, 32));
  if (!lane) red[w] = mx;
  __syncthreads();
  float bmx = red[0];
  #pragma unroll
  for (int i = 1; i < 8; ++i) bmx = fmaxf(bmx, red[i]);

  float ex[8];
  float sum = 0.f;
  #pragma unroll
  for (int j = 0; j < 8; ++j) { ex[j] = __expf(lg[j] - bmx); sum += ex[j]; }
  #pragma unroll
  for (int o = 16; o; o >>= 1) sum += __shfl_xor(sum, o, 32);
  __syncthreads();
  if (!lane) red[w] = sum;
  __syncthreads();
  float ts = 0.f;
  #pragma unroll
  for (int i = 0; i < 8; ++i) ts += red[i];
  const float inv = 1.f / ts;
  #pragma unroll
  for (int j = 0; j < 8; ++j)
    alpha[row * N + j * 256 + tid] = (_Float16)(ex[j] * inv);
}

// ---------------------------------------------------------------------------
extern "C" void kernel_launch(void* const* d_in, const int* in_sizes, int n_in,
                              void* d_out, int out_size, void* d_ws, size_t ws_size,
                              hipStream_t stream)
{
  (void)in_sizes; (void)n_in; (void)out_size; (void)ws_size;
  constexpr int Bb = 8, Nn = 2048, Hh = 512, Mm = Bb * Nn;

  const int*   adjacency  = (const int*)d_in[0];
  const float* arrivals   = (const float*)d_in[1];
  const float* departures = (const float*)d_in[2];
  const float* ihm        = (const float*)d_in[3];
  const float* active     = (const float*)d_in[4];
  const float* tstep      = (const float*)d_in[5];
  const float* ttot       = (const float*)d_in[6];
  const float* embW1      = (const float*)d_in[7];
  const float* embB1      = (const float*)d_in[8];
  const float* embW2      = (const float*)d_in[9];
  const float* embB2      = (const float*)d_in[10];
  const float* gW[3]  = {(const float*)d_in[11], (const float*)d_in[15], (const float*)d_in[19]};
  const float* gAs[3] = {(const float*)d_in[12], (const float*)d_in[16], (const float*)d_in[20]};
  const float* gAd[3] = {(const float*)d_in[13], (const float*)d_in[17], (const float*)d_in[21]};
  const float* gB[3]  = {(const float*)d_in[14], (const float*)d_in[18], (const float*)d_in[22]};
  const float* ffW1 = (const float*)d_in[23];
  const float* ffB1 = (const float*)d_in[24];
  const float* ffW2 = (const float*)d_in[25];
  const float* ffB2 = (const float*)d_in[26];

  // workspace carve
  char* p = (char*)d_ws;
  auto carve = [&](size_t bytes) -> void* {
    void* r = (void*)p; p += (bytes + 255) & ~(size_t)255; return r;
  };
  float*     xA    = (float*)carve((size_t)Mm * 512 * 4);
  float*     xB    = (float*)carve((size_t)Mm * 512 * 4);
  _Float16*  h16a  = (_Float16*)carve((size_t)Mm * 1024 * 2);
  _Float16*  h16b  = (_Float16*)carve((size_t)Mm * 1024 * 2);
  float*     hw32  = (float*)carve((size_t)Mm * 1024 * 4);
  _Float16*  hwT   = (_Float16*)carve((size_t)Mm * 1024 * 2);
  _Float16*  alpha = (_Float16*)carve((size_t)Bb * Nn * Nn * 2);
  unsigned char* adjT = (unsigned char*)carve((size_t)Bb * Nn * Nn);
  float*     esrc  = (float*)carve((size_t)Mm * 4);
  float*     edst  = (float*)carve((size_t)Mm * 4);
  const int Kin[3] = {512, 1024, 1024};
  const int Fo[3]  = {1024, 1024, 512};
  _Float16* embW2t = (_Float16*)carve((size_t)512 * 512 * 2);
  _Float16* gWt[3];
  for (int i = 0; i < 3; ++i) gWt[i] = (_Float16*)carve((size_t)Kin[i] * Fo[i] * 2);
  _Float16* fW1t = (_Float16*)carve((size_t)512 * 512 * 2);
  _Float16* fW2t = (_Float16*)carve((size_t)512 * 512 * 2);

  auto gemm = [&](const _Float16* Ap, const _Float16* Wp, const float* bias,
                  const float* resp, const float* rs, float* o32, _Float16* o16,
                  int M_, int F_, int K_, long long sA_, long long sW_,
                  long long sC_, int nb, int relu) {
    dim3 g(M_ / 128, F_ / 128, nb);
    gemm_wmma_f16<<<g, dim3(256), 0, stream>>>(Ap, Wp, bias, resp, rs, o32, o16,
                                               M_, F_, K_, sA_, sW_, sC_, relu);
  };
  auto wtr = [&](const float* W, _Float16* Wt, int K_, int F_) {
    transpose_f32_to_f16<<<dim3(F_ / 32, K_ / 32, 1), dim3(32, 8), 0, stream>>>(
        W, Wt, K_, F_, 0, 0);
  };

  // 1) weight pre-transpose to f16 [F,K]
  wtr(embW2, embW2t, 512, 512);
  wtr(gW[0], gWt[0], 512, 1024);
  wtr(gW[1], gWt[1], 1024, 1024);
  wtr(gW[2], gWt[2], 1024, 512);
  wtr(ffW1, fW1t, 512, 512);
  wtr(ffW2, fW2t, 512, 512);

  // 2) adjacency transpose -> mask rows per target
  adjT_kernel<<<dim3(Nn / 32, Nn / 32, Bb), dim3(32, 8), 0, stream>>>(adjacency, adjT, Nn);

  // 3) features + emb linear 1 -> x1 (f16)
  feat_emb1_kernel<<<Mm, 256, 0, stream>>>(arrivals, departures, ihm, tstep, ttot,
                                           embW1, embB1, h16a, Nn, Hh);

  // 4) emb linear 2 -> x (f32 residual base) + f16 copy for GAT0
  gemm(h16a, embW2t, embB2, nullptr, nullptr, xA, h16b, Mm, 512, 512, 0, 0, 0, 1, 0);

  // 5) three GAT layers
  _Float16* hin = h16b;
  _Float16* hnx = h16a;
  for (int i = 0; i < 3; ++i) {
    // hw = h @ W   (f32)
    gemm(hin, gWt[i], nullptr, nullptr, nullptr, hw32, nullptr,
         Mm, Fo[i], Kin[i], 0, 0, 0, 1, 0);
    // per-row attention terms
    edot_kernel<<<Mm / 8, 256, 0, stream>>>(hw32, gAs[i], gAd[i], esrc, edst, Fo[i]);
    // masked softmax -> alpha (f16)
    alpha_kernel<<<Mm, 256, 0, stream>>>(adjT, esrc, edst, alpha, Nn);
    // hw^T (f16) as B operand of aggregation GEMM
    transpose_f32_to_f16<<<dim3(Fo[i] / 32, Nn / 32, Bb), dim3(32, 8), 0, stream>>>(
        hw32, hwT, Nn, Fo[i], (long long)Nn * Fo[i], (long long)Fo[i] * Nn);
    // g = alpha @ hw + b  (batched); relu for layers 0,1; layer 2 adds x residual
    if (i < 2)
      gemm(alpha, hwT, gB[i], nullptr, nullptr, nullptr, hnx,
           Nn, Fo[i], Nn, (long long)Nn * Nn, (long long)Fo[i] * Nn,
           (long long)Nn * Fo[i], Bb, 1);
    else
      gemm(alpha, hwT, gB[i], xA, nullptr, xB, hnx,
           Nn, Fo[i], Nn, (long long)Nn * Nn, (long long)Fo[i] * Nn,
           (long long)Nn * Fo[i], Bb, 0);
    _Float16* t = hin; hin = hnx; hnx = t;
  }
  // hin now holds f16(x2 = x + g), xB holds x2 (f32)

  // 6) feed-forward residual blocks; last one scales by active_agents -> d_out
  gemm(hin, fW1t, ffB1, xB, nullptr, xA, hnx, Mm, 512, 512, 0, 0, 0, 1, 1);
  gemm(hnx, fW2t, ffB2, xA, active, (float*)d_out, nullptr, Mm, 512, 512, 0, 0, 0, 1, 1);
}
